// SelfMatcher_24352464569884
// MI455X (gfx1250) — matrix operand order repaired and checked by reading, not compiled, not run on hardware
//
#include <hip/hip_runtime.h>
#include <hip/hip_bf16.h>

// Problem constants (reference: L=512, B=32, D=H=256)
static constexpr int LQ = 512;
static constexpr int BQ = 32;
static constexpr int DQ = 256;   // == H
static constexpr int H3 = 768;   // 3*H

typedef __attribute__((ext_vector_type(8)))  __bf16 v8bf;
typedef __attribute__((ext_vector_type(16))) __bf16 v16bf;
typedef __attribute__((ext_vector_type(8)))  float  v8f;

// ---------------- fast transcendentals (v_exp_f32 / v_rcp_f32) ----------------
__device__ __forceinline__ float fast_tanh(float x) {
  // tanh(x) = 1 - 2/(1+exp(2x)); exp(2x) = exp2(x * 2*log2(e))
  float e = __builtin_amdgcn_exp2f(x * 2.885390081777927f);
  return 1.0f - 2.0f * __builtin_amdgcn_rcpf(1.0f + e);
}
__device__ __forceinline__ float fast_sigmoid(float x) {
  float e = __builtin_amdgcn_exp2f(-x * 1.4426950408889634f);
  return __builtin_amdgcn_rcpf(1.0f + e);
}

// ---------------- WMMA tile helpers (ISA 7.12.2 layouts, wave32) ----------------
// A-matrix 16x32 bf16 (row-major source): lane&15 = M row, lane>>4 = K-offset
// half. Per lane the 16 elems are two contiguous 8-wide K runs:
//   e=0..7  -> K = 8*half + e ;  e=8..15 -> K = 16 + 8*half + (e-8)
__device__ __forceinline__ v16bf load_a_tile(const __bf16* base, int lda) {
  int lane = threadIdx.x & 31;
  const __bf16* row = base + (size_t)(lane & 15) * lda + ((lane >> 4) << 3);
  v8bf lo = *(const v8bf*)(row);
  v8bf hi = *(const v8bf*)(row + 16);
  return __builtin_shufflevector(lo, hi, 0,1,2,3,4,5,6,7,8,9,10,11,12,13,14,15);
}

// B operand: pre-swizzled blocks. A 32(K)x16(N) tile occupies 512 contiguous
// bf16: element (k,n) at [lane*16 + e], lane = ((k>>3)&1)<<4 | (n&15), e = (k&7) | (((k>>4)&1)<<3).
// So each lane just loads 32 contiguous bytes.
__device__ __forceinline__ size_t sw_index(int k, int n, int Ntiles16) {
  int lane = (((k >> 3) & 1) << 4) | (n & 15);
  int e = (k & 7) | (((k >> 4) & 1) << 3);
  return ((size_t)(k >> 5) * Ntiles16 + (n >> 4)) * 512 + (size_t)lane * 16 + e;
}

// C/D 16x16 f32: lane&15 = N, VGPR r holds M = r + 8*(lane>>4)
__device__ __forceinline__ void store_c_bf16(__bf16* base, int ldc, v8f c) {
  int lane = threadIdx.x & 31;
  __bf16* p = base + (size_t)((lane >> 4) * 8) * ldc + (lane & 15);
#pragma unroll
  for (int r = 0; r < 8; ++r) p[(size_t)r * ldc] = (__bf16)c[r];
}
__device__ __forceinline__ void store_c_f32(float* base, int ldc, v8f c) {
  int lane = threadIdx.x & 31;
  float* p = base + (size_t)((lane >> 4) * 8) * ldc + (lane & 15);
#pragma unroll
  for (int r = 0; r < 8; ++r) p[(size_t)r * ldc] = c[r];
}
__device__ __forceinline__ void store_c_f32_bias(float* base, int ldc, v8f c,
                                                 const float* biasN) {
  int lane = threadIdx.x & 31;
  int n = lane & 15;
  float bv = biasN[n];
  float* p = base + (size_t)((lane >> 4) * 8) * ldc + n;
#pragma unroll
  for (int r = 0; r < 8; ++r) p[(size_t)r * ldc] = c[r] + bv;
}

// ---------------- conversion / swizzle kernels ----------------
__global__ void k_f32_to_bf16(const float* __restrict__ in, __bf16* __restrict__ out, int n) {
  int i = blockIdx.x * blockDim.x + threadIdx.x;
  if (i < n) out[i] = (__bf16)in[i];
}

// Swizzle an f32 weight W[n, k] (row-major, N x K) into WMMA-B blocks for the
// K x N logical operand: B(k,n) = W[n*K + k].
__global__ void k_swizzleB_f32T(const float* __restrict__ in, __bf16* __restrict__ out,
                                int K, int N) {
  int i = blockIdx.x * blockDim.x + threadIdx.x;
  if (i >= K * N) return;
  int k = i % K, n = i / K;                 // i = n*K + k -> coalesced reads
  out[sw_index(k, n, N >> 4)] = (__bf16)in[i];
}

// Swizzle a (batched) bf16 matrix: B_b(k,n) = src[b*sb + k*sk + n]
__global__ void k_swizzleB_bf16(const __bf16* __restrict__ in, __bf16* __restrict__ out,
                                int K, int N, long sk, long sb, long outStrideB) {
  int b = blockIdx.y;
  int i = blockIdx.x * blockDim.x + threadIdx.x;
  if (i >= K * N) return;
  int n = i % N, k = i / N;
  out[(size_t)b * outStrideB + sw_index(k, n, N >> 4)] =
      in[(size_t)b * sb + (size_t)k * sk + n];
}

// ---------------- GEMM: C[M,N] = A_bf16[M,K] * Bsw_bf16[K,N] ----------------
// One wave computes a 32(M) x 64(N) strip (8 WMMA tiles, B reused across the
// two M rows -> 12 b128 loads per 8 WMMAs). Block = 8 waves. Optional batch
// dim in blockIdx.y with per-batch element strides. B is pre-swizzled.
__global__ __launch_bounds__(256)
void k_gemm_bf16(const __bf16* __restrict__ A, int lda, long sA,
                 const __bf16* __restrict__ Bsw, int Ntiles16, long sB,
                 __bf16* __restrict__ C, int ldc, long sC,
                 int M, int K, int wavesPerRow) {
  long bb = blockIdx.y;
  A += bb * sA; Bsw += bb * sB; C += bb * sC;
  int w  = blockIdx.x * (blockDim.x >> 5) + (threadIdx.x >> 5);
  int mt = w / wavesPerRow, nt = w % wavesPerRow;   // mt: 32-row strip
  if (mt * 32 >= M) return;
  const __bf16* Arow0 = A + (size_t)(mt * 32) * lda;
  const __bf16* Arow1 = Arow0 + (size_t)16 * lda;
  int lane = threadIdx.x & 31;
  v8f c00 = {}, c01 = {}, c02 = {}, c03 = {};
  v8f c10 = {}, c11 = {}, c12 = {}, c13 = {};
  for (int k = 0; k < K; k += 32) {
    v16bf a0 = load_a_tile(Arow0 + k, lda);
    v16bf a1 = load_a_tile(Arow1 + k, lda);
    const __bf16* blk = Bsw + ((size_t)(k >> 5) * Ntiles16 + nt * 4) * 512 + (size_t)lane * 16;
    v16bf b0 = *(const v16bf*)(blk);
    v16bf b1 = *(const v16bf*)(blk + 512);
    v16bf b2 = *(const v16bf*)(blk + 1024);
    v16bf b3 = *(const v16bf*)(blk + 1536);
    __builtin_prefetch(blk + (size_t)Ntiles16 * 512, 0, 1);   // next k-tile
    c00 = __builtin_amdgcn_wmma_f32_16x16x32_bf16(false, a0, false, b0, (short)0, c00, false, false);
    c10 = __builtin_amdgcn_wmma_f32_16x16x32_bf16(false, a1, false, b0, (short)0, c10, false, false);
    c01 = __builtin_amdgcn_wmma_f32_16x16x32_bf16(false, a0, false, b1, (short)0, c01, false, false);
    c11 = __builtin_amdgcn_wmma_f32_16x16x32_bf16(false, a1, false, b1, (short)0, c11, false, false);
    c02 = __builtin_amdgcn_wmma_f32_16x16x32_bf16(false, a0, false, b2, (short)0, c02, false, false);
    c12 = __builtin_amdgcn_wmma_f32_16x16x32_bf16(false, a1, false, b2, (short)0, c12, false, false);
    c03 = __builtin_amdgcn_wmma_f32_16x16x32_bf16(false, a0, false, b3, (short)0, c03, false, false);
    c13 = __builtin_amdgcn_wmma_f32_16x16x32_bf16(false, a1, false, b3, (short)0, c13, false, false);
  }
  __bf16* Crow0 = C + (size_t)(mt * 32) * ldc + nt * 64;
  __bf16* Crow1 = Crow0 + (size_t)16 * ldc;
  store_c_bf16(Crow0 +  0, ldc, c00);
  store_c_bf16(Crow0 + 16, ldc, c01);
  store_c_bf16(Crow0 + 32, ldc, c02);
  store_c_bf16(Crow0 + 48, ldc, c03);
  store_c_bf16(Crow1 +  0, ldc, c10);
  store_c_bf16(Crow1 + 16, ldc, c11);
  store_c_bf16(Crow1 + 32, ldc, c12);
  store_c_bf16(Crow1 + 48, ldc, c13);
}

// GEMM with f32 output + bias (for gi = ctx @ W_ih^T + b_ih)
__global__ __launch_bounds__(256)
void k_gemm_bf16_bias_f32(const __bf16* __restrict__ A, int lda,
                          const __bf16* __restrict__ Bsw, int Ntiles16,
                          float* __restrict__ C, int ldc,
                          const float* __restrict__ bias,
                          int M, int K, int wavesPerRow) {
  int w  = blockIdx.x * (blockDim.x >> 5) + (threadIdx.x >> 5);
  int mt = w / wavesPerRow, nt = w % wavesPerRow;
  if (mt * 32 >= M) return;
  const __bf16* Arow0 = A + (size_t)(mt * 32) * lda;
  const __bf16* Arow1 = Arow0 + (size_t)16 * lda;
  int lane = threadIdx.x & 31;
  v8f c00 = {}, c01 = {}, c02 = {}, c03 = {};
  v8f c10 = {}, c11 = {}, c12 = {}, c13 = {};
  for (int k = 0; k < K; k += 32) {
    v16bf a0 = load_a_tile(Arow0 + k, lda);
    v16bf a1 = load_a_tile(Arow1 + k, lda);
    const __bf16* blk = Bsw + ((size_t)(k >> 5) * Ntiles16 + nt * 4) * 512 + (size_t)lane * 16;
    v16bf b0 = *(const v16bf*)(blk);
    v16bf b1 = *(const v16bf*)(blk + 512);
    v16bf b2 = *(const v16bf*)(blk + 1024);
    v16bf b3 = *(const v16bf*)(blk + 1536);
    __builtin_prefetch(blk + (size_t)Ntiles16 * 512, 0, 1);
    c00 = __builtin_amdgcn_wmma_f32_16x16x32_bf16(false, a0, false, b0, (short)0, c00, false, false);
    c10 = __builtin_amdgcn_wmma_f32_16x16x32_bf16(false, a1, false, b0, (short)0, c10, false, false);
    c01 = __builtin_amdgcn_wmma_f32_16x16x32_bf16(false, a0, false, b1, (short)0, c01, false, false);
    c11 = __builtin_amdgcn_wmma_f32_16x16x32_bf16(false, a1, false, b1, (short)0, c11, false, false);
    c02 = __builtin_amdgcn_wmma_f32_16x16x32_bf16(false, a0, false, b2, (short)0, c02, false, false);
    c12 = __builtin_amdgcn_wmma_f32_16x16x32_bf16(false, a1, false, b2, (short)0, c12, false, false);
    c03 = __builtin_amdgcn_wmma_f32_16x16x32_bf16(false, a0, false, b3, (short)0, c03, false, false);
    c13 = __builtin_amdgcn_wmma_f32_16x16x32_bf16(false, a1, false, b3, (short)0, c13, false, false);
  }
  float* Crow0 = C + (size_t)(mt * 32) * ldc + nt * 64;
  float* Crow1 = Crow0 + (size_t)16 * ldc;
  const float* bN = bias + nt * 64;
  store_c_f32_bias(Crow0 +  0, ldc, c00, bN +  0);
  store_c_f32_bias(Crow0 + 16, ldc, c01, bN + 16);
  store_c_f32_bias(Crow0 + 32, ldc, c02, bN + 32);
  store_c_f32_bias(Crow0 + 48, ldc, c03, bN + 48);
  store_c_f32_bias(Crow1 +  0, ldc, c10, bN +  0);
  store_c_f32_bias(Crow1 + 16, ldc, c11, bN + 16);
  store_c_f32_bias(Crow1 + 32, ldc, c12, bN + 32);
  store_c_f32_bias(Crow1 + 48, ldc, c13, bN + 48);
}

// ---------------- scores + softmax ----------------
// s[i,b,l] = sum_h tanh(xq[i,b,h] + xv[l,b,h]) * Vvec[b,h];  a = softmax_l(s)
// Block = one (i,b) row, 512 threads (thread = l). Output bf16 A_b[i,l], b-major.
__global__ __launch_bounds__(512)
void k_score_softmax(const __bf16* __restrict__ xq, const __bf16* __restrict__ xv,
                     const float* __restrict__ Vvec, __bf16* __restrict__ a_out) {
  int i = blockIdx.x, b = blockIdx.y, t = threadIdx.x;  // t == l
  __shared__ float xq_s[DQ];
  __shared__ float vv_s[DQ];
  __shared__ float red[LQ];
  if (t < DQ) {
    xq_s[t] = (float)xq[((size_t)i * BQ + b) * DQ + t];
    vv_s[t] = Vvec[b * DQ + t];
  }
  __syncthreads();
  const v8bf* xvp = (const v8bf*)(xv + ((size_t)t * BQ + b) * DQ);
  float acc = 0.f;
#pragma unroll 4
  for (int h8 = 0; h8 < DQ / 8; ++h8) {
    v8bf xvv = xvp[h8];
#pragma unroll
    for (int j = 0; j < 8; ++j) {
      int h = h8 * 8 + j;
      acc += fast_tanh(xq_s[h] + (float)xvv[j]) * vv_s[h];
    }
  }
  // block softmax over 512 lanes
  red[t] = acc;
  __syncthreads();
  for (int s = LQ / 2; s > 0; s >>= 1) {
    if (t < s) red[t] = fmaxf(red[t], red[t + s]);
    __syncthreads();
  }
  float mx = red[0];
  __syncthreads();
  float e = __builtin_amdgcn_exp2f((acc - mx) * 1.4426950408889634f);
  red[t] = e;
  __syncthreads();
  for (int s = LQ / 2; s > 0; s >>= 1) {
    if (t < s) red[t] += red[t + s];
    __syncthreads();
  }
  float inv = __builtin_amdgcn_rcpf(red[0]);
  a_out[(size_t)b * LQ * LQ + (size_t)i * LQ + t] = (__bf16)(e * inv);
}

// ---------------- sequential GRU scan ----------------
// One persistent workgroup (32 waves). h lives in LDS (f32 + bf16 mirror).
// Per step: gh = h_bf(32x256) @ WhhT(256x768). Wave w owns A-row mt = w>>4 and
// 3 consecutive N tiles, so one A load (LDS) feeds 3 WMMAs per k-step.
__global__ __launch_bounds__(1024)
void k_gru_scan(const float* __restrict__ gi, const __bf16* __restrict__ WhhSw,
                const float* __restrict__ b_hh, const float* __restrict__ h0,
                float* __restrict__ out) {
  extern __shared__ char smem[];
  __bf16* h_bf = (__bf16*)smem;                                  // 32*256*2  = 16 KB
  float*  h_f  = (float*)(smem + BQ * DQ * 2);                   // 32*256*4  = 32 KB
  float*  gh   = (float*)(smem + BQ * DQ * 2 + BQ * DQ * 4);     // 32*768*4  = 96 KB
  int tid = threadIdx.x;
  int wave = tid >> 5;
  int lane = tid & 31;
  int mt = wave >> 4;              // 0..1  (two 16-row strips of h)
  int nt0 = (wave & 15) * 3;       // 3 consecutive 16-col tiles of gh
#pragma unroll
  for (int k = 0; k < 8; ++k) {    // 8192 h elements
    int e = tid + k * 1024;
    float v = h0[e];
    h_f[e] = v;
    h_bf[e] = (__bf16)v;
  }
  __syncthreads();
  for (int step = 0; step < LQ; ++step) {
    // ---- gh = h @ WhhT : 2 (M) x 48 (N) tiles; 3 N-tiles per wave
    v8f acc0 = {}, acc1 = {}, acc2 = {};
    for (int k = 0; k < DQ; k += 32) {
      v16bf a = load_a_tile(h_bf + (size_t)(mt * 16) * DQ + k, DQ);
      const __bf16* blk = WhhSw + ((size_t)(k >> 5) * 48 + nt0) * 512 + (size_t)lane * 16;
      v16bf b0 = *(const v16bf*)(blk);
      v16bf b1 = *(const v16bf*)(blk + 512);
      v16bf b2 = *(const v16bf*)(blk + 1024);
      acc0 = __builtin_amdgcn_wmma_f32_16x16x32_bf16(false, a, false, b0, (short)0, acc0, false, false);
      acc1 = __builtin_amdgcn_wmma_f32_16x16x32_bf16(false, a, false, b1, (short)0, acc1, false, false);
      acc2 = __builtin_amdgcn_wmma_f32_16x16x32_bf16(false, a, false, b2, (short)0, acc2, false, false);
    }
    float* ghRow = gh + (size_t)(mt * 16) * H3 + nt0 * 16;
    store_c_f32(ghRow +  0, H3, acc0);
    store_c_f32(ghRow + 16, H3, acc1);
    store_c_f32(ghRow + 32, H3, acc2);
    __syncthreads();
    // ---- gates + state update
    const float* gis = gi + (size_t)step * BQ * H3;
#pragma unroll
    for (int k = 0; k < 8; ++k) {
      int e = tid + k * 1024;             // e = b*256 + h
      int b = e >> 8, h = e & 255;
      const float* gib = gis + b * H3;
      const float* ghb = gh + b * H3;
      float r = fast_sigmoid(gib[h]       + ghb[h]       + b_hh[h]);
      float z = fast_sigmoid(gib[256 + h] + ghb[256 + h] + b_hh[256 + h]);
      float n = fast_tanh(gib[512 + h] + r * (ghb[512 + h] + b_hh[512 + h]));
      float hn = (1.f - z) * n + z * h_f[e];
      h_f[e] = hn;
      h_bf[e] = (__bf16)hn;
      out[(size_t)step * (BQ * DQ) + e] = hn;
    }
    __syncthreads();
  }
}

// ---------------- host launcher ----------------
extern "C" void kernel_launch(void* const* d_in, const int* in_sizes, int n_in,
                              void* d_out, int out_size, void* d_ws, size_t ws_size,
                              hipStream_t stream) {
  (void)in_sizes; (void)n_in; (void)out_size; (void)ws_size;
  const float* v    = (const float*)d_in[0];   // [L,B,D]
  const float* h0   = (const float*)d_in[1];   // [B,H]
  const float* Vvec = (const float*)d_in[2];   // [B,H]
  const float* Wp   = (const float*)d_in[3];   // [H,D]
  const float* Wp_  = (const float*)d_in[4];   // [H,D]
  const float* Wih  = (const float*)d_in[5];   // [3H,D]
  const float* Whh  = (const float*)d_in[6];   // [3H,H]
  const float* bih  = (const float*)d_in[7];   // [3H]
  const float* bhh  = (const float*)d_in[8];   // [3H]
  float* out = (float*)d_out;                  // hs [L,B,H]

  char* ws = (char*)d_ws;
  size_t off = 0;
  auto carve = [&](size_t bytes) -> void* {
    void* p = ws + off;
    off = (off + bytes + 255) & ~(size_t)255;
    return p;
  };
  const size_t NV = (size_t)LQ * BQ * DQ;                    // 4,194,304
  const size_t VSW_B = (size_t)(LQ / 32) * (DQ / 16) * 512;  // per-b swizzled v: 131072
  __bf16* vb     = (__bf16*)carve(NV * 2);                   // v in bf16 (GEMM-A + swizzle src)
  __bf16* WpSw   = (__bf16*)carve((size_t)DQ * DQ * 2);      // Wp^T  swizzled (K=D, N=H)
  __bf16* Wp2Sw  = (__bf16*)carve((size_t)DQ * DQ * 2);
  __bf16* WihSw  = (__bf16*)carve((size_t)DQ * H3 * 2);      // W_ih^T swizzled (K=D, N=3H)
  __bf16* WhhSw  = (__bf16*)carve((size_t)DQ * H3 * 2);      // W_hh^T swizzled
  __bf16* vSw    = (__bf16*)carve((size_t)BQ * VSW_B * 2);   // per-b v as WMMA-B (K=L, N=D)
  __bf16* xq_bf  = (__bf16*)carve(NV * 2);                   // [L*B, H]
  __bf16* xv_bf  = (__bf16*)carve(NV * 2);                   // [L*B, H]
  __bf16* a_bf   = (__bf16*)carve((size_t)BQ * LQ * LQ * 2); // attn weights, b-major
  __bf16* ctx_bf = (__bf16*)carve(NV * 2);                   // context [L*B, D]
  float*  gi     = (float*)carve((size_t)LQ * BQ * H3 * 4);  // gi [L*B, 3H]

  // 1. conversions / swizzles
  k_f32_to_bf16<<<(int)((NV + 255) / 256), 256, 0, stream>>>(v, vb, (int)NV);
  k_swizzleB_f32T<<<(DQ * DQ + 255) / 256, 256, 0, stream>>>(Wp,  WpSw,  DQ, DQ);
  k_swizzleB_f32T<<<(DQ * DQ + 255) / 256, 256, 0, stream>>>(Wp_, Wp2Sw, DQ, DQ);
  k_swizzleB_f32T<<<(DQ * H3 + 255) / 256, 256, 0, stream>>>(Wih, WihSw, DQ, H3);
  k_swizzleB_f32T<<<(DQ * H3 + 255) / 256, 256, 0, stream>>>(Whh, WhhSw, DQ, H3);
  // per-b v as B operand: B_b(k=l, n=d) = vb[(l*B + b)*D + d]
  k_swizzleB_bf16<<<dim3((LQ * DQ + 255) / 256, BQ), 256, 0, stream>>>(
      vb, vSw, LQ, DQ, (long)BQ * DQ, (long)DQ, (long)VSW_B);

  // 2. projections: xq = v @ Wp^T, xv = v @ Wp_^T   (M=16384, K=256, N=256)
  {
    int M = LQ * BQ, wpr = DQ / 64;                 // wavesPerRow = 4
    int waves = (M / 32) * wpr;                     // 2048
    dim3 grid(waves / 8, 1);
    k_gemm_bf16<<<grid, 256, 0, stream>>>(vb, DQ, 0, WpSw,  DQ / 16, 0, xq_bf, DQ, 0, M, DQ, wpr);
    k_gemm_bf16<<<grid, 256, 0, stream>>>(vb, DQ, 0, Wp2Sw, DQ / 16, 0, xv_bf, DQ, 0, M, DQ, wpr);
  }

  // 3. scores + softmax  (grid: i x b)
  k_score_softmax<<<dim3(LQ, BQ), LQ, 0, stream>>>(xq_bf, xv_bf, Vvec, a_bf);

  // 4. context: per b, C_b[i,d] = A_b(512x512) @ v_b(512x256); C stored [i*B+b, d]
  {
    int M = LQ, wpr = DQ / 64;                      // 4
    int waves = (M / 32) * wpr;                     // 64
    dim3 grid(waves / 8, BQ);
    k_gemm_bf16<<<grid, 256, 0, stream>>>(a_bf, LQ, (long)LQ * LQ,
                                          vSw, DQ / 16, (long)VSW_B,
                                          ctx_bf, BQ * DQ, (long)DQ,
                                          M, LQ, wpr);
  }

  // 5. gi = ctx @ W_ih^T + b_ih   (M=16384, K=256, N=768)
  {
    int M = LQ * BQ, wpr = H3 / 64;                 // 12
    int waves = (M / 32) * wpr;                     // 6144
    dim3 grid(waves / 8, 1);
    k_gemm_bf16_bias_f32<<<grid, 256, 0, stream>>>(ctx_bf, DQ, WihSw, H3 / 16, gi, H3, bih, M, DQ, wpr);
  }

  // 6. sequential GRU scan (one persistent workgroup; h + gh in LDS)
  {
    size_t lds = (size_t)BQ * DQ * 2 + (size_t)BQ * DQ * 4 + (size_t)BQ * H3 * 4; // 144 KB
    k_gru_scan<<<1, 1024, lds, stream>>>(gi, WhhSw, bhh, h0, out);
  }
}
